// ScaledDotProductAttention_62792421868300
// MI455X (gfx1250) — matrix operand order repaired and checked by reading, not compiled, not run on hardware
//
#include <hip/hip_runtime.h>

typedef __attribute__((ext_vector_type(16))) _Float16 v16h;
typedef __attribute__((ext_vector_type(8)))  _Float16 v8h;
typedef __attribute__((ext_vector_type(4)))  _Float16 v4h;
typedef __attribute__((ext_vector_type(8)))  float    v8f;
typedef __attribute__((ext_vector_type(4)))  float    v4f;

#define LDH 72   // f16 LDS leading dim: 16B-aligned rows
#define LDF 68   // f32 LDS leading dim: 16B-aligned rows

// ---------- WMMA wrapper: D = A(16x32 f16) * B(32x16 f16) + C(f32) ----------
__device__ __forceinline__ v8f wmma16(v16h a, v16h b, v8f c) {
  return __builtin_amdgcn_wmma_f32_16x16x32_f16(
      /*neg_a=*/false, a, /*neg_b=*/false, b,
      /*c_mod=*/(short)0, c, /*reuse_a=*/false, /*reuse_b=*/false);
}

__device__ __forceinline__ v16h cat16(v8h lo, v8h hi) {
  return __builtin_shufflevector(lo, hi, 0, 1, 2, 3, 4, 5, 6, 7,
                                 8, 9, 10, 11, 12, 13, 14, 15);
}

// A fragment from row-major [m][k] storage: two 16B ds_load_b128.
__device__ __forceinline__ v16h frag_a(const _Float16* M, int m0, int k0,
                                       int lo16, int hi8) {
  const _Float16* p = M + (m0 + lo16) * LDH + k0 + 2 * hi8;
  v8h lo = *(const v8h*)p;
  v8h hi = *(const v8h*)(p + 8);
  return cat16(lo, hi);
}

// B fragment where logical B[k][n] = M[n][k]: two 16B ds_load_b128.
__device__ __forceinline__ v16h frag_bt(const _Float16* M, int k0, int n0,
                                        int lo16, int hi16) {
  const _Float16* p = M + (n0 + lo16) * LDH + k0 + hi16;
  v8h lo = *(const v8h*)p;
  v8h hi = *(const v8h*)(p + 8);
  return cat16(lo, hi);
}

// =====================================================================
// Kernel 1: attn_variable[bh][d][e] = softmax_e( sum_n (k[n,d]/8) v[n,e] )
// =====================================================================
__global__ void __launch_bounds__(256)
attn_variable_kernel(const float* __restrict__ K, const float* __restrict__ V,
                     float* __restrict__ attn_var) {
  __shared__ __align__(16) _Float16 Kt[64 * LDH];  // Kt[d][n] = k[n][d]/8
  __shared__ __align__(16) _Float16 Vt[64 * LDH];  // Vt[e][n] = v[n][e]
  __shared__ __align__(16) float Sf[64 * LDF];
  __shared__ float rowm[64], rowinv[64];

  const int bh = blockIdx.x;
  const int tid = threadIdx.x;
  const int lane = tid & 31, wave = tid >> 5;
  const int lo16 = lane & 15, hi = lane >> 4;
  const int hi8 = hi * 8, hi16 = hi * 16;
  const int mi = wave >> 1, njb = (wave & 1) * 2;
  const long base = (long)bh * 2048 * 64;

  v8f acc0 = {}; v8f acc1 = {};
  for (int n0 = 0; n0 < 2048; n0 += 64) {
    // batched vector loads, then convert + transpose-store
    v4f tk[4], tv[4];
#pragma unroll
    for (int i = 0; i < 4; ++i) {
      int idx4 = tid + i * 256;
      tk[i] = *(const v4f*)(K + base + (long)n0 * 64 + idx4 * 4);
      tv[i] = *(const v4f*)(V + base + (long)n0 * 64 + idx4 * 4);
    }
    if (n0 + 64 < 2048) {
      __builtin_prefetch(K + base + (long)(n0 + 64) * 64 + tid * 16, 0, 1);
      __builtin_prefetch(V + base + (long)(n0 + 64) * 64 + tid * 16, 0, 1);
    }
#pragma unroll
    for (int i = 0; i < 4; ++i) {
      int idx4 = tid + i * 256;
      int n = idx4 >> 4, d0 = (idx4 & 15) * 4;
#pragma unroll
      for (int e = 0; e < 4; ++e) {
        Kt[(d0 + e) * LDH + n] = (_Float16)(tk[i][e] * 0.125f);
        Vt[(d0 + e) * LDH + n] = (_Float16)(tv[i][e]);
      }
    }
    __syncthreads();
#pragma unroll
    for (int kc = 0; kc < 64; kc += 32) {
      v16h a  = frag_a(Kt, mi * 16, kc, lo16, hi8);          // A[d][n]
      v16h b0 = frag_bt(Vt, kc, njb * 16, lo16, hi16);       // B[n][e]=Vt[e][n]
      v16h b1 = frag_bt(Vt, kc, (njb + 1) * 16, lo16, hi16);
      acc0 = wmma16(a, b0, acc0);
      acc1 = wmma16(a, b1, acc1);
    }
    __syncthreads();
  }
#pragma unroll
  for (int r = 0; r < 8; ++r) {
    int row = mi * 16 + r + 8 * hi;
    Sf[row * LDF + njb * 16 + lo16]       = acc0[r];
    Sf[row * LDF + (njb + 1) * 16 + lo16] = acc1[r];
  }
  __syncthreads();
  if (tid < 64) {
    float m = -1e30f;
    for (int c = 0; c < 64; ++c) m = fmaxf(m, Sf[tid * LDF + c]);
    float l = 0.f;
    for (int c = 0; c < 64; ++c) l += __expf(Sf[tid * LDF + c] - m);
    rowm[tid] = m;
    rowinv[tid] = 1.0f / l;
  }
  __syncthreads();
#pragma unroll
  for (int i = 0; i < 4; ++i) {
    int idx4 = tid + i * 256;
    int r = idx4 >> 4, c0 = (idx4 & 15) * 4;
    v4f o;
#pragma unroll
    for (int e = 0; e < 4; ++e)
      o[e] = __expf(Sf[r * LDF + c0 + e] - rowm[r]) * rowinv[r];
    *(v4f*)(attn_var + (long)bh * 4096 + r * 64 + c0) = o;
  }
}

// =====================================================================
// Kernel 2: per (b,h, 64-row Q block):
//   output_variable = Q @ AV ; flash-style 2-pass softmax of S = (Q/8)K^T;
//   pass2 writes P to attn_time and accumulates O1 = P @ V
// =====================================================================
__global__ void __launch_bounds__(256)
attention_kernel(const float* __restrict__ Q, const float* __restrict__ K,
                 const float* __restrict__ V, const float* __restrict__ AV,
                 float* __restrict__ attn_time, float* __restrict__ out_time,
                 float* __restrict__ out_var) {
  __shared__ __align__(16) _Float16 Qh[64 * LDH];   // Qh[row][d] (scaled 1/8)
  __shared__ __align__(16) _Float16 Kh[64 * LDH];   // Kh[key][d]
  __shared__ __align__(16) _Float16 Vt[64 * LDH];   // Vt[d][key] = v[key][d]
  __shared__ __align__(16) _Float16 Ph[64 * LDH];   // Ph[row][key]; pass1: pm/ps
  __shared__ __align__(16) _Float16 AVt[64 * LDH];  // AVt[e][d] = AV[d][e]
  __shared__ __align__(16) float Sf[64 * LDF];
  __shared__ float rowM[64], rowL[64], rowInv[64], rowMn[64];

  float* pm = (float*)Ph;        // [64][4] partial max   (pass1 only)
  float* ps = pm + 256;          // [64][4] partial sums  (pass1 only)

  const int bh = blockIdx.y;
  const int rb = blockIdx.x;
  const int row0g = rb * 64;
  const int tid = threadIdx.x;
  const int lane = tid & 31, wave = tid >> 5;
  const int lo16 = lane & 15, hi = lane >> 4;
  const int hi8 = hi * 8, hi16 = hi * 16;
  const int mi = wave >> 1, njb = (wave & 1) * 2;
  const long base = (long)bh * 2048 * 64;

  // ---- stage Q (scaled 1/8, row-major) and AV^T ----
  {
    v4f tq[4], ta[4];
#pragma unroll
    for (int i = 0; i < 4; ++i) {
      int idx4 = tid + i * 256;
      tq[i] = *(const v4f*)(Q + base + (long)row0g * 64 + idx4 * 4);
      ta[i] = *(const v4f*)(AV + (long)bh * 4096 + idx4 * 4);
    }
#pragma unroll
    for (int i = 0; i < 4; ++i) {
      int idx4 = tid + i * 256;
      int r = idx4 >> 4, c0 = (idx4 & 15) * 4;
      v4h q4;
#pragma unroll
      for (int e = 0; e < 4; ++e) {
        q4[e] = (_Float16)(tq[i][e] * 0.125f);
        AVt[(c0 + e) * LDH + r] = (_Float16)(ta[i][e]);  // AVt[e][d]
      }
      *(v4h*)(Qh + r * LDH + c0) = q4;
    }
  }
  if (tid < 64) { rowM[tid] = -1e30f; rowL[tid] = 0.f; }
  __syncthreads();

  // ---- output_variable = Q @ AV (result * 8), staged through Sf ----
  {
    v8f o0 = {}; v8f o1 = {};
#pragma unroll
    for (int kc = 0; kc < 64; kc += 32) {
      v16h a  = frag_a(Qh, mi * 16, kc, lo16, hi8);
      v16h b0 = frag_bt(AVt, kc, njb * 16, lo16, hi16);
      v16h b1 = frag_bt(AVt, kc, (njb + 1) * 16, lo16, hi16);
      o0 = wmma16(a, b0, o0);
      o1 = wmma16(a, b1, o1);
    }
#pragma unroll
    for (int r = 0; r < 8; ++r) {
      int row = mi * 16 + r + 8 * hi;
      Sf[row * LDF + njb * 16 + lo16]       = o0[r] * 8.0f;
      Sf[row * LDF + (njb + 1) * 16 + lo16] = o1[r] * 8.0f;
    }
    __syncthreads();
#pragma unroll
    for (int i = 0; i < 4; ++i) {
      int idx4 = tid + i * 256;
      int r = idx4 >> 4, c0 = (idx4 & 15) * 4;
      v4f t = *(const v4f*)(Sf + r * LDF + c0);
      *(v4f*)(out_var + base + (long)(row0g + r) * 64 + c0) = t;
    }
    __syncthreads();
  }

  // ---- pass 1: online row max / sumexp over 32 key blocks ----
  for (int j = 0; j < 32; ++j) {
    v4f tk[4];
#pragma unroll
    for (int i = 0; i < 4; ++i)
      tk[i] = *(const v4f*)(K + base + (long)j * 4096 + (tid + i * 256) * 4);
    if (j + 1 < 32)
      __builtin_prefetch(K + base + (long)(j + 1) * 4096 + tid * 16, 0, 1);
#pragma unroll
    for (int i = 0; i < 4; ++i) {
      int idx4 = tid + i * 256;
      int r = idx4 >> 4, c0 = (idx4 & 15) * 4;
      v4h k4;
#pragma unroll
      for (int e = 0; e < 4; ++e) k4[e] = (_Float16)(tk[i][e]);
      *(v4h*)(Kh + r * LDH + c0) = k4;
    }
    __syncthreads();
    v8f s0 = {}; v8f s1 = {};
#pragma unroll
    for (int kc = 0; kc < 64; kc += 32) {
      v16h a  = frag_a(Qh, mi * 16, kc, lo16, hi8);
      v16h b0 = frag_bt(Kh, kc, njb * 16, lo16, hi16);
      v16h b1 = frag_bt(Kh, kc, (njb + 1) * 16, lo16, hi16);
      s0 = wmma16(a, b0, s0);
      s1 = wmma16(a, b1, s1);
    }
#pragma unroll
    for (int r = 0; r < 8; ++r) {
      int row = mi * 16 + r + 8 * hi;
      Sf[row * LDF + njb * 16 + lo16]       = s0[r];
      Sf[row * LDF + (njb + 1) * 16 + lo16] = s1[r];
    }
    __syncthreads();
    {  // 4-way parallel row scan: max
      int r = tid >> 2, qd = tid & 3;
      const float* rp = Sf + r * LDF + qd * 16;
      float bm = rp[0];
#pragma unroll
      for (int c = 1; c < 16; ++c) bm = fmaxf(bm, rp[c]);
      pm[r * 4 + qd] = bm;
    }
    __syncthreads();
    if (tid < 64) {
      float bm = fmaxf(fmaxf(pm[tid * 4 + 0], pm[tid * 4 + 1]),
                       fmaxf(pm[tid * 4 + 2], pm[tid * 4 + 3]));
      rowMn[tid] = fmaxf(rowM[tid], bm);
    }
    __syncthreads();
    {  // 4-way parallel row scan: sum of exp
      int r = tid >> 2, qd = tid & 3;
      const float* rp = Sf + r * LDF + qd * 16;
      float mN = rowMn[r], s = 0.f;
#pragma unroll
      for (int c = 0; c < 16; ++c) s += __expf(rp[c] - mN);
      ps[r * 4 + qd] = s;
    }
    __syncthreads();
    if (tid < 64) {
      float sum = (ps[tid * 4 + 0] + ps[tid * 4 + 1]) +
                  (ps[tid * 4 + 2] + ps[tid * 4 + 3]);
      float mOld = rowM[tid], mNew = rowMn[tid];
      rowL[tid] = rowL[tid] * __expf(mOld - mNew) + sum;
      rowM[tid] = mNew;
    }
    __syncthreads();
  }
  if (tid < 64) rowInv[tid] = 1.0f / rowL[tid];
  __syncthreads();

  // ---- pass 2: recompute S, emit P (global + LDS f16), O1 += P @ V ----
  v8f o10 = {}; v8f o11 = {};
  for (int j = 0; j < 32; ++j) {
    v4f tk[4], tv[4];
#pragma unroll
    for (int i = 0; i < 4; ++i) {
      int idx4 = tid + i * 256;
      tk[i] = *(const v4f*)(K + base + (long)j * 4096 + idx4 * 4);
      tv[i] = *(const v4f*)(V + base + (long)j * 4096 + idx4 * 4);
    }
    if (j + 1 < 32) {
      __builtin_prefetch(K + base + (long)(j + 1) * 4096 + tid * 16, 0, 1);
      __builtin_prefetch(V + base + (long)(j + 1) * 4096 + tid * 16, 0, 1);
    }
#pragma unroll
    for (int i = 0; i < 4; ++i) {
      int idx4 = tid + i * 256;
      int r = idx4 >> 4, c0 = (idx4 & 15) * 4;
      v4h k4;
#pragma unroll
      for (int e = 0; e < 4; ++e) {
        k4[e] = (_Float16)(tk[i][e]);
        Vt[(c0 + e) * LDH + r] = (_Float16)(tv[i][e]);  // Vt[d][key]
      }
      *(v4h*)(Kh + r * LDH + c0) = k4;
    }
    __syncthreads();
    v8f s0 = {}; v8f s1 = {};
#pragma unroll
    for (int kc = 0; kc < 64; kc += 32) {
      v16h a  = frag_a(Qh, mi * 16, kc, lo16, hi8);
      v16h b0 = frag_bt(Kh, kc, njb * 16, lo16, hi16);
      v16h b1 = frag_bt(Kh, kc, (njb + 1) * 16, lo16, hi16);
      s0 = wmma16(a, b0, s0);
      s1 = wmma16(a, b1, s1);
    }
#pragma unroll
    for (int r = 0; r < 8; ++r) {
      int row = mi * 16 + r + 8 * hi;
      float p0 = __expf(s0[r] - rowM[row]) * rowInv[row];
      float p1 = __expf(s1[r] - rowM[row]) * rowInv[row];
      Ph[row * LDH + njb * 16 + lo16]       = (_Float16)p0;
      Ph[row * LDH + (njb + 1) * 16 + lo16] = (_Float16)p1;
      Sf[row * LDF + njb * 16 + lo16]       = p0;
      Sf[row * LDF + (njb + 1) * 16 + lo16] = p1;
    }
    __syncthreads();  // Ph + Sf complete
    // coalesced f32 P write from Sf
    float* at = attn_time + ((long)bh * 2048 + row0g) * 2048 + (long)j * 64;
#pragma unroll
    for (int i = 0; i < 4; ++i) {
      int idx4 = tid + i * 256;
      int r = idx4 >> 4, c0 = (idx4 & 15) * 4;
      v4f pv = *(const v4f*)(Sf + r * LDF + c0);
      *(v4f*)(at + (long)r * 2048 + c0) = pv;
    }
    // O1 += P @ V
#pragma unroll
    for (int kc = 0; kc < 64; kc += 32) {
      v16h a  = frag_a(Ph, mi * 16, kc, lo16, hi8);
      v16h b0 = frag_bt(Vt, kc, njb * 16, lo16, hi16);
      v16h b1 = frag_bt(Vt, kc, (njb + 1) * 16, lo16, hi16);
      o10 = wmma16(a, b0, o10);
      o11 = wmma16(a, b1, o11);
    }
    __syncthreads();  // safe to overwrite Kh/Vt/Ph/Sf next iteration
  }

  // ---- write output_time, staged through Sf for coalesced b128 stores ----
#pragma unroll
  for (int r = 0; r < 8; ++r) {
    int row = mi * 16 + r + 8 * hi;
    Sf[row * LDF + njb * 16 + lo16]       = o10[r];
    Sf[row * LDF + (njb + 1) * 16 + lo16] = o11[r];
  }
  __syncthreads();
#pragma unroll
  for (int i = 0; i < 4; ++i) {
    int idx4 = tid + i * 256;
    int r = idx4 >> 4, c0 = (idx4 & 15) * 4;
    v4f t = *(const v4f*)(Sf + r * LDF + c0);
    *(v4f*)(out_time + base + (long)(row0g + r) * 64 + c0) = t;
  }
}

// =====================================================================
extern "C" void kernel_launch(void* const* d_in, const int* in_sizes, int n_in,
                              void* d_out, int out_size, void* d_ws, size_t ws_size,
                              hipStream_t stream) {
  const float* q = (const float*)d_in[0];
  const float* k = (const float*)d_in[1];
  const float* v = (const float*)d_in[2];
  // d_in[3] = epoch (unused: reference branch inactive for epoch <= 10)

  float* out = (float*)d_out;
  const long SZ_OT = 4L * 8 * 2048 * 64;     // 4,194,304
  const long SZ_AT = 4L * 8 * 2048 * 2048;   // 134,217,728
  float* out_time  = out;
  float* out_var   = out + SZ_OT;
  float* attn_time = out + 2 * SZ_OT;
  float* attn_var  = out + 2 * SZ_OT + SZ_AT;

  attn_variable_kernel<<<dim3(32), dim3(256), 0, stream>>>(k, v, attn_var);
  attention_kernel<<<dim3(32, 32), dim3(256), 0, stream>>>(
      q, k, v, attn_var, attn_time, out_time, out_var);
}
